// Head_CompleteInformation_70952859730359
// MI455X (gfx1250) — matrix operand order repaired and checked by reading, not compiled, not run on hardware
//
#include <hip/hip_runtime.h>
#include <hip/hip_bf16.h>

// ---------------------------------------------------------------------------
// Single-head attention (B=8, T=2048, E=1024, H=64) for MI455X / gfx1250.
//   k = x Wk^T ; q = x Wq^T ; v = x Wv^T       (f16 WMMA, f32 accum)
//   out = softmax(q k^T / 8) v                 (flash-attention, online softmax)
//
// Round-2 changes vs round 1:
//  * qkv projection fused: one wave per 16-row x tile computes all 12 output
//    tiles (3 mats x 4 col-tiles) -> x A-fragments loaded once, 12 WMMAs per
//    32-K step (was 1), ~12x less A-fragment VMEM traffic.
//  * flash attention widened to 64 keys/iter: 16 WMMAs per iter, half the
//    shuffle-reduction + O-rescale work per key.
// Workspace: xh(32MB) + Wh(384KB) + qh/kh/vT(2MB each) ~= 38.4 MB.
// ---------------------------------------------------------------------------

typedef _Float16 half_t;
typedef __attribute__((ext_vector_type(4)))  _Float16 v4h;
typedef __attribute__((ext_vector_type(8)))  _Float16 v8h;
typedef __attribute__((ext_vector_type(16))) _Float16 v16h;
typedef __attribute__((ext_vector_type(8)))  float    v8f;

#define N_EMBD 1024
#define HEAD   64
#define BATCH  8
#define SEQ    2048

static __device__ __forceinline__ v16h cat16(v8h lo, v8h hi) {
  return __builtin_shufflevector(lo, hi, 0,1,2,3,4,5,6,7,8,9,10,11,12,13,14,15);
}

static __device__ __forceinline__ v8f wmma_f16(v16h a, v16h b, v8f c) {
  // D = A(16x32 f16) * B(32x16 f16) + C(16x16 f32)
  return __builtin_amdgcn_wmma_f32_16x16x32_f16(false, a, false, b,
                                                (short)0, c, false, false);
}

// ---------------------------------------------------------------------------
// fp32 -> f16 conversion, 4-wide
// ---------------------------------------------------------------------------
__global__ void cvt_f32_f16x4(const float* __restrict__ src,
                              half_t* __restrict__ dst, int n4) {
  int i = blockIdx.x * blockDim.x + threadIdx.x;
  int stride = gridDim.x * blockDim.x;
  for (; i < n4; i += stride) {
    float4 f = ((const float4*)src)[i];
    v4h h;
    h[0] = (half_t)f.x; h[1] = (half_t)f.y;
    h[2] = (half_t)f.z; h[3] = (half_t)f.w;
    ((v4h*)dst)[i] = h;
  }
}

// ---------------------------------------------------------------------------
// Fused QKV projection. One wave per 16-row tile of x (1024 waves), each
// computing all 12 16x16 output tiles (q/k/v x 4 head-col tiles).
// A (x tile, 16x32): lanes 0-15 row M=lane K 0..15,
// lanes 16-31 row M=lane-16 K 16..31.
// B (W^T tile, 32x16): lane n holds W[n0+n][e..e+15] (lo) / +16 (hi) -> one
// contiguous 32B load.  v is stored transposed vT[b][h][t] for the PV stage.
// ---------------------------------------------------------------------------
__global__ void __launch_bounds__(128)
qkv_wmma(const half_t* __restrict__ xh, const half_t* __restrict__ Wh,
         half_t* __restrict__ qh, half_t* __restrict__ kh,
         half_t* __restrict__ vT) {
  const int lane = threadIdx.x & 31;
  const int wid  = blockIdx.x * 4 + (threadIdx.x >> 5);  // 0..1023
  const int r0   = wid * 16;                             // row-tile base (B*T)
  const int lh   = lane & 15;
  const int hi   = lane >> 4;

  const half_t* arow = xh + (size_t)(r0 + lh) * N_EMBD + (hi ? 16 : 0);
  // per-lane base into W for its B-matrix column (row lh of W), both k-halves
  const half_t* brow = Wh + (size_t)lh * N_EMBD + (hi ? 16 : 0);

  v8f acc[12] = {};                                      // [mat*4 + ntile]
  for (int e = 0; e < N_EMBD; e += 32) {
    v16h a = cat16(*(const v8h*)(arow + e), *(const v8h*)(arow + e + 8));
#pragma unroll
    for (int mat = 0; mat < 3; ++mat) {
#pragma unroll
      for (int nt = 0; nt < 4; ++nt) {
        const half_t* bp = brow + ((size_t)mat * HEAD + nt * 16) * N_EMBD + e;
        acc[mat * 4 + nt] = wmma_f16(a, *(const v16h*)bp, acc[mat * 4 + nt]);
      }
    }
  }

  // C layout: lane holds column N=n0+lh; VGPR v -> row v (lo) / v+8 (hi)
#pragma unroll
  for (int nt = 0; nt < 4; ++nt) {
    const int col = nt * 16 + lh;
#pragma unroll
    for (int v = 0; v < 8; ++v) {
      int row = r0 + v + hi * 8;                         // over B*T
      qh[(size_t)row * HEAD + col] = (half_t)acc[0 * 4 + nt][v];
      kh[(size_t)row * HEAD + col] = (half_t)acc[1 * 4 + nt][v];
      int bb = row >> 11, t = row & (SEQ - 1);
      vT[((size_t)bb * HEAD + col) * SEQ + t] = (half_t)acc[2 * 4 + nt][v];
    }
  }
}

// ---------------------------------------------------------------------------
// Flash attention. One wave per 16-query tile (1024 waves, 4 waves/block).
// Per 64-key step: 8 WMMAs (S = Q K^T, four 16-key C tiles), online softmax
// (shuffle-reduce inside each 16-lane half = one C-matrix row group),
// P transposed C->A layout through LDS, then 8 WMMAs (O += P V).
// K/V per batch is 0.5 MB f16 -> resident in the 192 MB L2 across all waves.
// ---------------------------------------------------------------------------
__global__ void __launch_bounds__(128)
flash_attn(const half_t* __restrict__ qh, const half_t* __restrict__ kh,
           const half_t* __restrict__ vT, float* __restrict__ out) {
  __shared__ alignas(16) half_t pbuf[4][16][72];        // 72-half pitch = 144B
  const int lane = threadIdx.x & 31;
  const int warp = threadIdx.x >> 5;
  const int wid  = blockIdx.x * 4 + warp;               // 0..1023
  const int b    = wid >> 7;
  const int q0   = (wid & 127) * 16;
  const int lh   = lane & 15;
  const int hi   = lane >> 4;

  // Q fragments for head dims 0..31 / 32..63 (A layout)
  const half_t* qrow = qh + ((size_t)(b * SEQ + q0 + lh)) * HEAD + (hi ? 16 : 0);
  const v16h aq0 = cat16(*(const v8h*)(qrow +  0), *(const v8h*)(qrow +  8));
  const v16h aq1 = cat16(*(const v8h*)(qrow + 32), *(const v8h*)(qrow + 40));

  const half_t* kbase = kh + (size_t)b * SEQ * HEAD;
  const half_t* vbase = vT + (size_t)b * HEAD * SEQ;

  float m[8], l[8];
  v8f o[4] = {};
#pragma unroll
  for (int v = 0; v < 8; ++v) { m[v] = -1e30f; l[v] = 0.0f; }

  for (int jj = 0; jj < SEQ; jj += 64) {
    if (jj + 64 < SEQ) {  // uniform: pull next K/V tiles toward L2/L0
      __builtin_prefetch(kbase + (size_t)(jj + 64 + lane) * HEAD, 0, 1);
      __builtin_prefetch(vbase + (size_t)lane * SEQ + jj + 64, 0, 1);
    }

    // ---- S = Q K^T for 64 keys (four 16-key tiles) -----------------------
    v8f s[4];
#pragma unroll
    for (int g = 0; g < 4; ++g) {
      const half_t* kp = kbase + (size_t)(jj + g * 16 + lh) * HEAD + (hi ? 16 : 0);
      v8f c = {};
      c = wmma_f16(aq0, *(const v16h*)(kp),      c);   // h 0..31
      c = wmma_f16(aq1, *(const v16h*)(kp + 32), c);   // h 32..63
      s[g] = c;
    }

    // ---- online softmax over the 64 new keys -----------------------------
#pragma unroll
    for (int v = 0; v < 8; ++v) {
      float a0 = s[0][v] * 0.125f, a1 = s[1][v] * 0.125f;
      float a2 = s[2][v] * 0.125f, a3 = s[3][v] * 0.125f;
      float t = fmaxf(fmaxf(a0, a1), fmaxf(a2, a3));
      t = fmaxf(t, __shfl_xor(t, 1));                  // reduce 16-lane half
      t = fmaxf(t, __shfl_xor(t, 2));
      t = fmaxf(t, __shfl_xor(t, 4));
      t = fmaxf(t, __shfl_xor(t, 8));
      float mn    = fmaxf(m[v], t);
      float alpha = __expf(m[v] - mn);
      m[v] = mn;
      a0 = __expf(a0 - mn); a1 = __expf(a1 - mn);
      a2 = __expf(a2 - mn); a3 = __expf(a3 - mn);
      float rs = (a0 + a1) + (a2 + a3);
      rs += __shfl_xor(rs, 1);
      rs += __shfl_xor(rs, 2);
      rs += __shfl_xor(rs, 4);
      rs += __shfl_xor(rs, 8);
      l[v] = l[v] * alpha + rs;
      o[0][v] *= alpha; o[1][v] *= alpha; o[2][v] *= alpha; o[3][v] *= alpha;
      s[0][v] = a0; s[1][v] = a1; s[2][v] = a2; s[3][v] = a3;
    }

    // ---- transpose P: C layout (f32, col-per-lane) -> A layout (f16) -----
    {
      half_t (*pt)[72] = pbuf[warp];
      const int rb = hi * 8;
#pragma unroll
      for (int v = 0; v < 8; ++v) {
        pt[rb + v][lh]      = (half_t)s[0][v];   // keys jj+ 0..15
        pt[rb + v][16 + lh] = (half_t)s[1][v];   // keys jj+16..31
        pt[rb + v][32 + lh] = (half_t)s[2][v];   // keys jj+32..47
        pt[rb + v][48 + lh] = (half_t)s[3][v];   // keys jj+48..63
      }
      // same-wave LDS: DS ops are in-order; wait + memory clobber pins the
      // store->load ordering before the A-fragment reload.
      asm volatile("s_wait_dscnt 0" ::: "memory");
    }
    const half_t* pr0 = &pbuf[warp][lh][hi ? 16 : 0];
    v16h pa0 = cat16(*(const v8h*)(pr0),      *(const v8h*)(pr0 + 8));  // k 0..31
    v16h pa1 = cat16(*(const v8h*)(pr0 + 32), *(const v8h*)(pr0 + 40)); // k 32..63

    // ---- O += P (16x64) * V (64x64); B from vT -> contiguous 32B loads ---
#pragma unroll
    for (int nt = 0; nt < 4; ++nt) {
      const half_t* vp = vbase + (size_t)(nt * 16 + lh) * SEQ + jj + (hi ? 16 : 0);
      o[nt] = wmma_f16(pa0, *(const v16h*)(vp),      o[nt]);  // keys jj+ 0..31
      o[nt] = wmma_f16(pa1, *(const v16h*)(vp + 32), o[nt]);  // keys jj+32..63
    }
  }

  // ---- normalize and store fp32 output ----------------------------------
#pragma unroll
  for (int nt = 0; nt < 4; ++nt) {
#pragma unroll
    for (int v = 0; v < 8; ++v) {
      int row = q0 + v + hi * 8;
      int col = nt * 16 + lh;
      out[((size_t)(b * SEQ + row)) * HEAD + col] = o[nt][v] / l[v];
    }
  }
}

// ---------------------------------------------------------------------------
extern "C" void kernel_launch(void* const* d_in, const int* in_sizes, int n_in,
                              void* d_out, int out_size, void* d_ws, size_t ws_size,
                              hipStream_t stream) {
  const float* x  = (const float*)d_in[0];
  const float* Wk = (const float*)d_in[1];
  const float* Wq = (const float*)d_in[2];
  const float* Wv = (const float*)d_in[3];
  float* out = (float*)d_out;

  // workspace carve-up (halves); all sub-buffers stay >=128B aligned
  half_t* xh = (half_t*)d_ws;                                  // 16,777,216
  half_t* Wh = xh + (size_t)BATCH * SEQ * N_EMBD;              // 3*65,536 (q,k,v)
  half_t* qh = Wh + (size_t)3 * HEAD * N_EMBD;                 // 1,048,576
  half_t* kh = qh + (size_t)BATCH * SEQ * HEAD;                // 1,048,576
  half_t* vT = kh + (size_t)BATCH * SEQ * HEAD;                // 1,048,576 [b][h][t]

  cvt_f32_f16x4<<<4096, 256, 0, stream>>>(x,  xh, BATCH * SEQ * N_EMBD / 4);
  cvt_f32_f16x4<<<64,   256, 0, stream>>>(Wq, Wh,                     HEAD * N_EMBD / 4);
  cvt_f32_f16x4<<<64,   256, 0, stream>>>(Wk, Wh + HEAD * N_EMBD,     HEAD * N_EMBD / 4);
  cvt_f32_f16x4<<<64,   256, 0, stream>>>(Wv, Wh + 2 * HEAD * N_EMBD, HEAD * N_EMBD / 4);

  qkv_wmma<<<256, 128, 0, stream>>>(xh, Wh, qh, kh, vT);    // 1024 waves, 12 WMMA/step
  flash_attn<<<256, 128, 0, stream>>>(qh, kh, vT, out);     // 1024 waves, 16 WMMA/step
}
